// Quantizer_11630771438211
// MI455X (gfx1250) — matrix operand order repaired and checked by reading, not compiled, not run on hardware
//
#include <hip/hip_runtime.h>

#define NROWS 32768   // B*S
#define DIM   1024    // D
#define GV    640     // G*V
#define VSZ   320     // V
#define DSUB  512     // D/G
#define FEPS  1e-7f

typedef __attribute__((ext_vector_type(16))) __bf16 v16bf;
typedef __attribute__((ext_vector_type(8)))  float  v8f;

union FragU { unsigned int u[8]; uint4 q[2]; v16bf v; };

__device__ __forceinline__ unsigned short f2bf(float f) {
    unsigned int u = __float_as_uint(f);
    u += 0x7FFFu + ((u >> 16) & 1u);          // round-to-nearest-even
    return (unsigned short)(u >> 16);
}

// ---------------------------------------------------------------------------
// Kernel 1: logits = x @ W + b  (bf16 WMMA, fp32 accum), software-pipelined.
// Logits land in d_out rows (stride 1024, cols 0..639) and stay L2-resident
// for pass 2. Tiling: 256 thr = 8 waves, 128(M) x 128(N), BK = 32.
// ---------------------------------------------------------------------------
__global__ __launch_bounds__(256) void gemm_logits(
    const float* __restrict__ x, const float* __restrict__ W,
    const float* __restrict__ bias, float* __restrict__ out)
{
    __shared__ unsigned short As[128 * 32];   // As[m][k]
    __shared__ unsigned short Bs[128 * 32];   // Bs[n][k] = W[k0+k][nBase+n]

    const int tid  = threadIdx.x;
    const int wave = tid >> 5;
    const int lane = tid & 31;
    const int nBase = blockIdx.x * 128;       // 5 blocks over 640 cols
    const int mBase = blockIdx.y * 128;       // 256 blocks over 32768 rows

    v8f acc[8];
    const v8f vzero = {0.f,0.f,0.f,0.f,0.f,0.f,0.f,0.f};
    for (int i = 0; i < 8; ++i) acc[i] = vzero;

    const int mloc = lane & 15;
    const int half = lane >> 4;

    // per-thread tile-load coordinates (fixed across k-steps)
    int aRow[4], aK[4], bK[4], bN[4];
    const float* aPtr[4];
    const float* bPtr[4];
    for (int i = 0; i < 4; ++i) {
        int f = tid + i * 256;
        aRow[i] = f >> 3;                 // 0..127
        aK[i]   = (f & 7) << 2;           // 0..28
        bK[i]   = f >> 5;                 // 0..31
        bN[i]   = (f & 31) << 2;          // 0..124
        aPtr[i] = x + (size_t)(mBase + aRow[i]) * DIM + aK[i];
        bPtr[i] = W + (size_t)bK[i] * GV + nBase + bN[i];
    }

    // ---- prologue: stage k-step 0 into registers
    float4 ax[4], bw[4];
    for (int i = 0; i < 4; ++i) {
        ax[i] = *(const float4*)(aPtr[i]);
        bw[i] = *(const float4*)(bPtr[i]);
    }

    for (int ks = 0; ks < DIM / 32; ++ks) {
        // ---- convert staged registers -> LDS (bf16)
        for (int i = 0; i < 4; ++i) {
            ushort4 h = make_ushort4(f2bf(ax[i].x), f2bf(ax[i].y),
                                     f2bf(ax[i].z), f2bf(ax[i].w));
            *(ushort4*)(&As[aRow[i] * 32 + aK[i]]) = h;
            Bs[(bN[i] + 0) * 32 + bK[i]] = f2bf(bw[i].x);
            Bs[(bN[i] + 1) * 32 + bK[i]] = f2bf(bw[i].y);
            Bs[(bN[i] + 2) * 32 + bK[i]] = f2bf(bw[i].z);
            Bs[(bN[i] + 3) * 32 + bK[i]] = f2bf(bw[i].w);
        }
        __syncthreads();

        // ---- issue next k-step's global loads (overlap with WMMA below)
        if (ks + 1 < DIM / 32) {
            for (int i = 0; i < 4; ++i) {
                ax[i] = *(const float4*)(aPtr[i] + (ks + 1) * 32);
                bw[i] = *(const float4*)(bPtr[i] + (size_t)(ks + 1) * 32 * GV);
            }
        }

        // ---- A fragment (16x32 bf16): lanes 0-15 hold K0-7,K16-23;
        //      lanes 16-31 hold K8-15,K24-31
        FragU fa;
        const unsigned short* arow = &As[(wave * 16 + mloc) * 32];
        fa.q[0] = *(const uint4*)(arow + half * 8);
        fa.q[1] = *(const uint4*)(arow + 16 + half * 8);

        for (int nt = 0; nt < 8; ++nt) {
            // B fragment (32x16 bf16): same K order as A (K0-7,K16-23 / K8-15,K24-31)
            FragU fb;
            const unsigned short* brow = &Bs[(nt * 16 + mloc) * 32 + half * 8];
            fb.q[0] = *(const uint4*)(brow);
            fb.q[1] = *(const uint4*)(brow + 16);
            acc[nt] = __builtin_amdgcn_wmma_f32_16x16x32_bf16(
                false, fa.v, false, fb.v, (short)0, acc[nt], false, false);
        }
        __syncthreads();
    }

    // ---- epilogue: C layout VGPR r -> M = r + half*8, N = lane%16
    for (int nt = 0; nt < 8; ++nt) {
        int col = nBase + nt * 16 + mloc;
        float bb = bias[col];
        for (int r = 0; r < 8; ++r) {
            int row = mBase + wave * 16 + r + half * 8;
            out[(size_t)row * 1024 + col] = acc[nt][r] + bb;
        }
    }
}

// ---------------------------------------------------------------------------
// Kernel 2: one wave per row n: noisy argmax + clean softmax prob
// accumulation, then overwrite the row with gathered codebook vectors.
// ---------------------------------------------------------------------------
__global__ __launch_bounds__(256) void gumbel_vq(
    float* __restrict__ out, const float* __restrict__ gumbel,
    const float* __restrict__ codebooks, float* __restrict__ gacc)
{
    __shared__ float accs[GV];
    const int tid = threadIdx.x;
    for (int i = tid; i < GV; i += 256) accs[i] = 0.f;
    __syncthreads();

    const int wave = tid >> 5;
    const int lane = tid & 31;
    const int n = blockIdx.x * 8 + wave;

    int sel[2];
    for (int g = 0; g < 2; ++g) {
        const float* lg = out + (size_t)n * 1024 + g * VSZ;       // clean logits
        const float* gu = gumbel + ((size_t)n * 2 + g) * VSZ;
        float lv[10];
        float mx = -3.4e38f, bn = -3.4e38f;
        int bi = 0x7fffffff;
        for (int i = 0; i < 10; ++i) {
            int v = lane + i * 32;
            float l = lg[v];
            lv[i] = l;
            mx = fmaxf(mx, l);
            float nz = l + gu[v];                                 // TAU = 1
            if (nz > bn || (nz == bn && v < bi)) { bn = nz; bi = v; }
        }
        // wave32 butterfly reductions
        for (int off = 16; off >= 1; off >>= 1) {
            mx = fmaxf(mx, __shfl_xor(mx, off, 32));
            float on = __shfl_xor(bn, off, 32);
            int   oi = __shfl_xor(bi, off, 32);
            if (on > bn || (on == bn && oi < bi)) { bn = on; bi = oi; }
        }
        float ev[10], s = 0.f;
        for (int i = 0; i < 10; ++i) { ev[i] = __expf(lv[i] - mx); s += ev[i]; }
        for (int off = 16; off >= 1; off >>= 1) s += __shfl_xor(s, off, 32);
        float inv = 1.f / s;
        for (int i = 0; i < 10; ++i)
            atomicAdd(&accs[g * VSZ + lane + i * 32], ev[i] * inv);
        sel[g] = bi;
        // warm the selected codebook row (global_prefetch) while g=1 computes
        __builtin_prefetch(codebooks + ((size_t)g * VSZ + bi) * DSUB + lane * 16, 0, 3);
    }

    // all 640 logits of row n consumed -> safe to overwrite row with gather
    for (int g = 0; g < 2; ++g) {
        const float4* cb = (const float4*)(codebooks + ((size_t)g * VSZ + sel[g]) * DSUB);
        float4* dst = (float4*)(out + (size_t)n * 1024 + g * DSUB);
        for (int j = 0; j < 4; ++j)
            dst[lane + j * 32] = cb[lane + j * 32];
    }

    __syncthreads();
    for (int i = tid; i < GV; i += 256) atomicAdd(&gacc[i], accs[i]);
}

// ---------------------------------------------------------------------------
// Kernel 3: perplexity finalize -> d_out[N*D]
// ---------------------------------------------------------------------------
__global__ void perplexity_k(const float* __restrict__ gacc, float* __restrict__ outp)
{
    __shared__ float sh[GV];
    const int t = threadIdx.x;                // 320 threads
    const float invN = 1.0f / (float)NROWS;
    for (int g = 0; g < 2; ++g) {
        float a = gacc[g * VSZ + t] * invN;
        sh[g * VSZ + t] = a * __logf(a + FEPS);
    }
    __syncthreads();
    if (t == 0) {
        float p = 0.f;
        for (int g = 0; g < 2; ++g) {
            float h = 0.f;
            for (int v = 0; v < VSZ; ++v) h += sh[g * VSZ + v];
            p += __expf(-h);                  // exp(-sum p*log(p+eps))
        }
        outp[0] = 0.5f * p;
    }
}

__global__ void zero_acc(float* __restrict__ g) { g[threadIdx.x] = 0.f; }

// ---------------------------------------------------------------------------
extern "C" void kernel_launch(void* const* d_in, const int* in_sizes, int n_in,
                              void* d_out, int out_size, void* d_ws, size_t ws_size,
                              hipStream_t stream)
{
    const float* x    = (const float*)d_in[0];   // [B,S,D] fp32
    const float* W    = (const float*)d_in[1];   // [D, G*V]
    const float* bias = (const float*)d_in[2];   // [G*V]
    const float* cb   = (const float*)d_in[3];   // [G,V,D/G]
    const float* gm   = (const float*)d_in[4];   // [N*G, V]
    float* out  = (float*)d_out;
    float* gacc = (float*)d_ws;                  // 640 floats of scratch

    zero_acc<<<1, GV, 0, stream>>>(gacc);

    dim3 g1(GV / 128, NROWS / 128);              // (5, 256)
    gemm_logits<<<g1, 256, 0, stream>>>(x, W, bias, out);

    gumbel_vq<<<NROWS / 8, 256, 0, stream>>>(out, gm, cb, gacc);

    perplexity_k<<<1, VSZ, 0, stream>>>(gacc, out + (size_t)NROWS * DIM);
}
